// Generator_10428180594765
// MI455X (gfx1250) — compile-verified, matrix operand order checked by reading, not run on hardware
//
#include <hip/hip_runtime.h>
#include <hip/hip_bf16.h>
#include <stdint.h>

typedef _Float16 v16h __attribute__((ext_vector_type(16)));
typedef _Float16 v8h  __attribute__((ext_vector_type(8)));
typedef float    v8f  __attribute__((ext_vector_type(8)));
typedef unsigned int u32x4 __attribute__((ext_vector_type(4)));
typedef int      i32x8 __attribute__((ext_vector_type(8)));
typedef int      i32x4 __attribute__((ext_vector_type(4)));

#define BATCH   8
#define CHN     512
#define KDIM    4608            // CHN * 9
#define KBLK    32
#define KSTEPS  (KDIM / KBLK)   // 144
#define TILE_M  128
#define TILE_N  128
#define LDH     40              // padded LDS row stride in halves (80B = 20 DW; conflict-free)
#define SQRT2F  1.41421356237309515f
#define INV_SQRT_512  0.04419417382415922f
#define INV_SQRT_4608 0.01473139127471974f

#if __has_builtin(__builtin_amdgcn_tensor_load_to_lds) && __has_builtin(__builtin_amdgcn_s_wait_tensorcnt)
#define USE_TDM 1
#else
#define USE_TDM 0
#endif

// ---------------------------------------------------------------------------
// s[st][b][o] = style[b] . (mod_w[o] * 1/sqrt(512)) + mod_b[o],  st in {0,1,2}
// ---------------------------------------------------------------------------
__global__ __launch_bounds__(256) void style_kernel(
    const float* __restrict__ style,
    const float* __restrict__ m1w, const float* __restrict__ m1b,
    const float* __restrict__ m2w, const float* __restrict__ m2b,
    const float* __restrict__ m3w, const float* __restrict__ m3b,
    float* __restrict__ sout) {
  int idx = blockIdx.x * 256 + threadIdx.x;     // 0 .. 3*8*512-1
  int st = idx >> 12;
  int r  = idx & 4095;
  int b  = r >> 9, o = r & 511;
  const float* mw = (st == 0) ? m1w : ((st == 1) ? m2w : m3w);
  const float* mb = (st == 0) ? m1b : ((st == 1) ? m2b : m3b);
  const float* sv = style + b * 512;
  const float* wr = mw + o * 512;
  float acc = 0.f;
  for (int j = 0; j < 512; ++j) acc += sv[j] * wr[j];
  sout[idx] = acc * INV_SQRT_512 + mb[o];
}

// ---------------------------------------------------------------------------
// Modulate + demodulate 3x3 weights -> f16, layout [b][o][k], k = tap*512 + i
// One 256-thread block per (stage, b, o).
// ---------------------------------------------------------------------------
__global__ __launch_bounds__(256) void modw_kernel(
    const float* __restrict__ w1, const float* __restrict__ w2,
    const float* __restrict__ sbuf,
    _Float16* __restrict__ wb1, _Float16* __restrict__ wb2) {
  int id = blockIdx.x;                      // 0 .. 2*8*512-1
  int st = id >> 12;
  int bo = id & 4095;
  int b = bo >> 9, o = bo & 511;
  const float* w  = (st == 0) ? w1 : w2;
  const float* sv = sbuf + st * 4096 + b * 512;
  _Float16* outw = ((st == 0) ? wb1 : wb2) + ((size_t)b * CHN + o) * KDIM;
  int tid = threadIdx.x;
  float v[18];
  float ssq = 0.f;
#pragma unroll
  for (int p = 0; p < 18; ++p) {
    int k = tid + p * 256;                  // k = t*512 + i
    int i = k & 511, t = k >> 9;
    float val = w[o * KDIM + i * 9 + t] * sv[i] * INV_SQRT_4608;
    v[p] = val;
    ssq += val * val;
  }
  __shared__ float red[256];
  red[tid] = ssq;
  __syncthreads();
  for (int s = 128; s > 0; s >>= 1) {
    if (tid < s) red[tid] += red[tid + s];
    __syncthreads();
  }
  float d = rsqrtf(red[0] + 1e-8f);
#pragma unroll
  for (int p = 0; p < 18; ++p) {
    int k = tid + p * 256;
    outw[k] = (_Float16)(v[p] * d);
  }
}

// ToRGB weights: no demodulation, keep f32. wb3[(b*3+o)*512 + i]
__global__ __launch_bounds__(256) void modw3_kernel(
    const float* __restrict__ w3, const float* __restrict__ sbuf,
    float* __restrict__ wb3) {
  int idx = blockIdx.x * 256 + threadIdx.x;  // 0 .. 8*3*512-1
  int b = idx / 1536;
  int rem = idx - b * 1536;
  int o = rem >> 9, i = rem & 511;
  float s = sbuf[2 * 4096 + b * 512 + i];
  wb3[idx] = w3[o * 512 + i] * s * INV_SQRT_512;
}

__global__ __launch_bounds__(256) void f32_to_f16_kernel(
    const float* __restrict__ in, _Float16* __restrict__ out, int n) {
  int i = blockIdx.x * 256 + threadIdx.x;
  if (i < n) out[i] = (_Float16)in[i];
}

// ---------------------------------------------------------------------------
// Implicit-GEMM modulated conv via WMMA f16 -> f32.
//   mode 0: conv_transpose2(stride 2) on 32x32 -> 65x65 (zero-insertion im2col)
//   mode 1: 3x3 conv pad 1 on 64x64 -> 64x64
// C[o,n] = sum_k Wmod[b][o][k] * im2col(Act)[b][k][n],  K = 4608.
// Workgroup: 256 thr = 8 waves, tile 128x128; wave tile 32x64 (2x4 WMMAs).
// A tile fetched by TDM (tensor_load_to_lds, padded to stride-40 rows);
// B tile (im2col) gathered by all threads into LDS. Double buffered.
// ---------------------------------------------------------------------------
__global__ __launch_bounds__(256) void conv_gemm_wmma(
    const _Float16* __restrict__ Wmod,   // [B][CHN][KDIM]
    const _Float16* __restrict__ Act,    // [B][CHN][Hin][Hin]
    _Float16* __restrict__ Out,          // [B][CHN][Wout*Wout]
    int mode) {
  const int b  = blockIdx.z;
  const int m0 = blockIdx.y * TILE_M;
  const int n0 = blockIdx.x * TILE_N;
  const int Wout  = (mode == 0) ? 65 : 64;
  const int Ntot  = Wout * Wout;
  const int HinSq = (mode == 0) ? 1024 : 4096;
  const int tid  = threadIdx.x;
  const int lane = tid & 31;
  const int wv   = tid >> 5;
  const int wm   = (wv >> 1) << 5;   // wave M offset: 0,32,64,96
  const int wn   = (wv & 1) << 6;    // wave N offset: 0,64

  __shared__ _Float16 smA[2][TILE_M * LDH];
  __shared__ _Float16 smB[2][TILE_N * LDH];

  const _Float16* Wb = Wmod + ((size_t)b * CHN + m0) * KDIM;
  const _Float16* Ab = Act + (size_t)b * CHN * HinSq;

  v8f acc[2][4];
#pragma unroll
  for (int i = 0; i < 2; ++i)
#pragma unroll
    for (int j = 0; j < 4; ++j)
#pragma unroll
      for (int r = 0; r < 8; ++r) acc[i][j][r] = 0.f;

  // Per-thread im2col invariants: thread covers one n, k = kof + 2p.
  const int bn  = tid & 127;
  const int kof = tid >> 7;
  const int N = n0 + bn;
  const bool nval = (N < Ntot);
  int y = 0, x = 0;
  if (nval) { y = N / Wout; x = N - y * Wout; }

  _Float16 regB[16];

  auto loadB = [&](int kb) {
    int t = kb >> 9;                       // tap index (constant within K-block)
    int ky = t / 3, kx = t - ky * 3;
    int ibase = kb & 511;
#pragma unroll
    for (int p = 0; p < 16; ++p) {
      int k = kof + 2 * p;
      _Float16 v = (_Float16)0.f;
      if (nval) {
        int i = ibase + k;
        if (mode == 0) {
          // out65[y,x] += w[ky,kx] * u[y-ky, x-kx]; u nonzero at even coords
          int uy = y - ky, ux = x - kx;
          if (((uy | ux) >= 0) && !((uy | ux) & 1)) {
            int iy = uy >> 1, ix = ux >> 1;
            if (iy < 32 && ix < 32) v = Ab[i * 1024 + iy * 32 + ix];
          }
        } else {
          int ay = y + ky - 1, ax = x + kx - 1;
          if (ay >= 0 && ay < 64 && ax >= 0 && ax < 64)
            v = Ab[(i << 12) + (ay << 6) + ax];
        }
      }
      regB[p] = v;
    }
  };
  auto storeB = [&](int buf) {
#pragma unroll
    for (int p = 0; p < 16; ++p) {
      int k = kof + 2 * p;
      smB[buf][bn * LDH + k] = regB[p];
    }
  };

#if USE_TDM
  // 2D TDM tile: 128 rows x 32 halves, row stride KDIM halves, padded into LDS
  // at stride 40 halves via pad_interval=16DW / pad_amount=4DW.
  auto loadA_tdm = [&](int buf, int kb) {
    if (wv != 0) return;
    unsigned long long ga = (unsigned long long)(uintptr_t)(Wb + kb);
    unsigned lds = (unsigned)(uintptr_t)(&smA[buf][0]);
    u32x4 g0;
    g0[0] = 1u;                                        // count=1 valid descriptor
    g0[1] = lds;                                       // lds_addr
    g0[2] = (unsigned)ga;                              // global_addr[31:0]
    g0[3] = (unsigned)((ga >> 32) & 0x1FFFFFFu) | (2u << 30);  // addr[56:32] | type=2
    i32x8 g1;
    g1[0] = (1 << 16) | (1 << 20) | (3 << 22) | (3 << 25); // data_size=2B,pad_en,int=16DW,amt=4DW
    g1[1] = (int)((KDIM & 0xFFFF) << 16);              // tensor_dim0 lo16
    g1[2] = (int)((KDIM >> 16) | ((CHN & 0xFFFF) << 16)); // td0 hi | td1 lo
    g1[3] = (int)((CHN >> 16) | (KBLK << 16));         // td1 hi | tile_dim0=32
    g1[4] = TILE_M;                                    // tile_dim1=128, tile_dim2=0
    g1[5] = KDIM;                                      // tensor_dim0_stride lo32
    g1[6] = 0;
    g1[7] = 0;
    i32x4 g2; g2[0] = 1; g2[1] = 1; g2[2] = 0; g2[3] = 0;  // dim2=1, dim3=1
    i32x4 g3; g3[0] = 0; g3[1] = 0; g3[2] = 0; g3[3] = 0;
    i32x8 g4;                                          // extra group (6-arg variant)
    g4[0] = 0; g4[1] = 0; g4[2] = 0; g4[3] = 0;
    g4[4] = 0; g4[5] = 0; g4[6] = 0; g4[7] = 0;
    __builtin_amdgcn_tensor_load_to_lds(g0, g1, g2, g3, g4, 0);
  };
#else
  v8h regA[2];
  auto loadA = [&](int kb) {
#pragma unroll
    for (int p = 0; p < 2; ++p) {
      int c = tid + (p << 8);
      int row = c >> 2, seg = c & 3;
      regA[p] = *(const v8h*)(Wb + row * KDIM + kb + seg * 8);
    }
  };
  auto storeA = [&](int buf) {
#pragma unroll
    for (int p = 0; p < 2; ++p) {
      int c = tid + (p << 8);
      int row = c >> 2, seg = c & 3;
      *(v8h*)&smA[buf][row * LDH + seg * 8] = regA[p];
    }
  };
#endif

  // 16-bit A frag (16x32): lanes 0-15 row M, K 0..7 & 16..23; lanes 16-31 K 8..15 & 24..31
  auto fragA = [&](int buf, int mt) {
    int row = wm + (mt << 4) + (lane & 15);
    int kp = (lane >> 4) << 3;
    const _Float16* p = &smA[buf][row * LDH + kp];
    v8h lo = *(const v8h*)p;
    v8h hi = *(const v8h*)(p + 16);
    v16h r;
#pragma unroll
    for (int j = 0; j < 8; ++j) { r[j] = lo[j]; r[8 + j] = hi[j]; }
    return r;
  };
  auto fragB = [&](int buf, int nt) {   // B stored column-major in LDS: smB[n][k]
    int col = wn + (nt << 4) + (lane & 15);
    int kp = (lane >> 4) << 3;
    const _Float16* p = &smB[buf][col * LDH + kp];
    v8h lo = *(const v8h*)p;
    v8h hi = *(const v8h*)(p + 16);
    v16h r;
#pragma unroll
    for (int j = 0; j < 8; ++j) { r[j] = lo[j]; r[8 + j] = hi[j]; }
    return r;
  };
  auto compute = [&](int buf) {
    v16h a0 = fragA(buf, 0), a1 = fragA(buf, 1);
#pragma unroll
    for (int nt = 0; nt < 4; ++nt) {
      v16h bb = fragB(buf, nt);
      acc[0][nt] = __builtin_amdgcn_wmma_f32_16x16x32_f16(
          false, a0, false, bb, (short)0, acc[0][nt], false, false);
      acc[1][nt] = __builtin_amdgcn_wmma_f32_16x16x32_f16(
          false, a1, false, bb, (short)0, acc[1][nt], false, false);
    }
  };

  // prologue
  loadB(0);
#if USE_TDM
  loadA_tdm(0, 0);
#else
  loadA(0);
  storeA(0);
#endif
  storeB(0);
#if USE_TDM
  if (wv == 0) __builtin_amdgcn_s_wait_tensorcnt(0);
#endif
  __syncthreads();

  for (int kt = 0; kt < KSTEPS; ++kt) {
    int cur = kt & 1, nxt = cur ^ 1;
    if (kt + 1 < KSTEPS) {
#if USE_TDM
      loadA_tdm(nxt, (kt + 1) * KBLK);
#else
      loadA((kt + 1) * KBLK);
#endif
      loadB((kt + 1) * KBLK);
    }
    compute(cur);
    if (kt + 1 < KSTEPS) {
#if !USE_TDM
      storeA(nxt);
#endif
      storeB(nxt);
#if USE_TDM
      if (wv == 0) __builtin_amdgcn_s_wait_tensorcnt(0);
#endif
    }
    __syncthreads();
  }

  // Epilogue: C layout -> VGPR r: lanes 0-15 hold M=r, lanes 16-31 hold M=8+r
  const int ml = (lane >> 4) << 3;
  const int nl = lane & 15;
#pragma unroll
  for (int mt = 0; mt < 2; ++mt)
#pragma unroll
    for (int nt = 0; nt < 4; ++nt)
#pragma unroll
      for (int r = 0; r < 8; ++r) {
        int m = m0 + wm + (mt << 4) + r + ml;
        int n = n0 + wn + (nt << 4) + nl;
        if (n < Ntot)
          Out[((size_t)b * CHN + m) * Ntot + n] = (_Float16)acc[mt][nt][r];
      }
}

// ---------------------------------------------------------------------------
// 4x4 blur on the 65x65 conv-transpose output + noise + bias + fused lrelu
// ---------------------------------------------------------------------------
__global__ __launch_bounds__(256) void blur_noise_act_kernel(
    const _Float16* __restrict__ t65, const float* __restrict__ noise1,
    const float* __restrict__ bias1, const float* __restrict__ nw,
    _Float16* __restrict__ act1) {
  int idx = blockIdx.x * 256 + threadIdx.x;         // 8*512*64*64
  int x = idx & 63, y = (idx >> 6) & 63, c = (idx >> 12) & 511, b = idx >> 21;
  const float k1[4] = {0.125f, 0.375f, 0.375f, 0.125f};
  const _Float16* src = t65 + ((size_t)(b * 512 + c)) * 4225;
  float a = 0.f;
#pragma unroll
  for (int by = 0; by < 4; ++by) {
    int sy = y - 1 + by;
    if (sy < 0 || sy > 64) continue;
#pragma unroll
    for (int bx = 0; bx < 4; ++bx) {
      int sx = x - 1 + bx;
      if (sx < 0 || sx > 64) continue;
      a += (4.f * k1[by] * k1[bx]) * (float)src[sy * 65 + sx];
    }
  }
  a += bias1[c] + nw[0] * noise1[(b << 12) + (y << 6) + x];
  a = (a > 0.f ? a : 0.2f * a) * SQRT2F;
  act1[idx] = (_Float16)a;
}

__global__ __launch_bounds__(256) void post2_kernel(
    const _Float16* __restrict__ t2, const float* __restrict__ noise2,
    const float* __restrict__ bias2, const float* __restrict__ nw,
    _Float16* __restrict__ act2) {
  int idx = blockIdx.x * 256 + threadIdx.x;
  int x = idx & 63, y = (idx >> 6) & 63, c = (idx >> 12) & 511, b = idx >> 21;
  float a = (float)t2[idx] + bias2[c] + nw[0] * noise2[(b << 12) + (y << 6) + x];
  a = (a > 0.f ? a : 0.2f * a) * SQRT2F;
  act2[idx] = (_Float16)a;
}

// ---------------------------------------------------------------------------
// ToRGB (1x1 modulated conv) + bias3 + upsampled-blurred skip
// ---------------------------------------------------------------------------
__global__ __launch_bounds__(256) void rgb_kernel(
    const _Float16* __restrict__ act2, const float* __restrict__ wb3,
    const float* __restrict__ bias3, const float* __restrict__ skip,
    float* __restrict__ out) {
  int idx = blockIdx.x * 256 + threadIdx.x;   // 8 * 4096
  int b = idx >> 12;
  int pix = idx & 4095;
  int y = pix >> 6, x = pix & 63;
  const _Float16* ap = act2 + ((size_t)b << 21) + pix;
  const float* wp = wb3 + b * 1536;
  float a0 = 0.f, a1 = 0.f, a2 = 0.f;
  for (int i = 0; i < 512; ++i) {
    float a = (float)ap[(size_t)i << 12];
    a0 += wp[i] * a;
    a1 += wp[512 + i] * a;
    a2 += wp[1024 + i] * a;
  }
  const float k1[4] = {0.125f, 0.375f, 0.375f, 0.125f};
  float sk0 = 0.f, sk1 = 0.f, sk2 = 0.f;
#pragma unroll
  for (int by = 0; by < 4; ++by) {
    int sy = y + by - 2;
    if (sy < 0 || (sy & 1) || (sy >> 1) > 31) continue;
#pragma unroll
    for (int bx = 0; bx < 4; ++bx) {
      int sx = x + bx - 2;
      if (sx < 0 || (sx & 1) || (sx >> 1) > 31) continue;
      float wgt = 4.f * k1[by] * k1[bx];
      int off = (sy >> 1) * 32 + (sx >> 1);
      sk0 += wgt * skip[((b * 3 + 0) << 10) + off];
      sk1 += wgt * skip[((b * 3 + 1) << 10) + off];
      sk2 += wgt * skip[((b * 3 + 2) << 10) + off];
    }
  }
  out[((b * 3 + 0) << 12) + pix] = a0 + bias3[0] + sk0;
  out[((b * 3 + 1) << 12) + pix] = a1 + bias3[1] + sk1;
  out[((b * 3 + 2) << 12) + pix] = a2 + bias3[2] + sk2;
}

// ---------------------------------------------------------------------------
extern "C" void kernel_launch(void* const* d_in, const int* in_sizes, int n_in,
                              void* d_out, int out_size, void* d_ws, size_t ws_size,
                              hipStream_t stream) {
  (void)in_sizes; (void)n_in; (void)out_size; (void)ws_size;
  const float* x      = (const float*)d_in[0];
  const float* style  = (const float*)d_in[1];
  const float* skip   = (const float*)d_in[2];
  const float* noise1 = (const float*)d_in[3];
  const float* noise2 = (const float*)d_in[4];
  const float* w1     = (const float*)d_in[5];
  const float* m1w    = (const float*)d_in[6];
  const float* m1b    = (const float*)d_in[7];
  const float* b1     = (const float*)d_in[8];
  const float* n1w    = (const float*)d_in[9];
  const float* w2     = (const float*)d_in[10];
  const float* m2w    = (const float*)d_in[11];
  const float* m2b    = (const float*)d_in[12];
  const float* b2     = (const float*)d_in[13];
  const float* n2w    = (const float*)d_in[14];
  const float* w3     = (const float*)d_in[15];
  const float* m3w    = (const float*)d_in[16];
  const float* m3b    = (const float*)d_in[17];
  const float* b3     = (const float*)d_in[18];
  float* out = (float*)d_out;

  // Workspace layout (lifetime-aliased):
  const size_t WB = (size_t)BATCH * CHN * KDIM * 2;     // 37,748,736 B per stage
  const size_t XH = (size_t)BATCH * CHN * 1024 * 2;     // 8,388,608 B
  char* ws = (char*)d_ws;
  float*    sbuf  = (float*)ws;                          // 3*8*512 f32
  float*    wb3   = (float*)(ws + 65536);                // 8*3*512  f32
  _Float16* wb1h  = (_Float16*)(ws + 131072);
  _Float16* wb2h  = (_Float16*)(ws + 131072 + WB);
  _Float16* xh    = (_Float16*)(ws + 131072 + 2 * WB);
  _Float16* t65   = (_Float16*)(ws + 131072 + 2 * WB + XH);
  _Float16* act1h = wb1h;   // alias: wb1h dead after stage-1 GEMM
  _Float16* t2h   = t65;    // alias: t65 dead after blur
  _Float16* act2h = wb2h;   // alias: wb2h dead after stage-2 GEMM

  style_kernel<<<48, 256, 0, stream>>>(style, m1w, m1b, m2w, m2b, m3w, m3b, sbuf);
  modw_kernel<<<8192, 256, 0, stream>>>(w1, w2, sbuf, wb1h, wb2h);
  modw3_kernel<<<48, 256, 0, stream>>>(w3, sbuf, wb3);
  f32_to_f16_kernel<<<16384, 256, 0, stream>>>(x, xh, BATCH * CHN * 1024);

  // Stage 1: conv_transpose as implicit GEMM (N = 65*65 -> 34 tiles)
  conv_gemm_wmma<<<dim3(34, 4, BATCH), 256, 0, stream>>>(wb1h, xh, t65, 0);
  blur_noise_act_kernel<<<65536, 256, 0, stream>>>(t65, noise1, b1, n1w, act1h);

  // Stage 2: 3x3 conv as implicit GEMM (N = 64*64 -> 32 tiles)
  conv_gemm_wmma<<<dim3(32, 4, BATCH), 256, 0, stream>>>(wb2h, act1h, t2h, 1);
  post2_kernel<<<65536, 256, 0, stream>>>(t2h, noise2, b2, n2w, act2h);

  // Stage 3: ToRGB + skip
  rgb_kernel<<<128, 256, 0, stream>>>(act2h, wb3, b3, skip, out);
}
